// Model_69544110457109
// MI455X (gfx1250) — matrix operand order repaired and checked by reading, not compiled, not run on hardware
//
#include <hip/hip_runtime.h>
#include <hip/hip_bf16.h>

typedef __attribute__((ext_vector_type(16))) _Float16 v16h;
typedef __attribute__((ext_vector_type(8)))  _Float16 v8h;
typedef __attribute__((ext_vector_type(8)))  float    v8f;

#define N_NODES 4096
#define F_IN    256
#define NHID    128
#define NHEADS  8
#define OUTDIM  256
#define HID_ALL (NHID*NHEADS)   // 1024
#define GAT_ALPHA 0.2f
#define LOG2E   1.4426950408889634f

// ---------------------------------------------------------------------------
// WMMA wrapper: D = A(16x32 f16) * B(32x16 f16) + C(16x16 f32)
// A-frag (ISA 7.12.2): lane&15 = row M, kh=lane>>4; elem e -> K = 8*kh+e (+8 if e>=8)
// B-frag: lane&15 = col N, kh=lane>>4; elem e -> K = 8*kh+e (+8 if e>=8)
// C-frag: vgpr r, lane: row = r + 8*(lane>>4), col = lane&15
// ---------------------------------------------------------------------------
__device__ __forceinline__ v8f wmma_f16(v16h a, v16h b, v8f c) {
    return __builtin_amdgcn_wmma_f32_16x16x32_f16(
        /*neg_a=*/false, a, /*neg_b=*/false, b,
        /*c_mod=*/(short)0, c, /*reuse_a=*/false, /*reuse_b=*/false);
}

// ---------------------------------------------------------------------------
// Prep: x [4096,256] f32 row-major -> A-fragment-swizzled f16
// layout: ((rt*8 + kt)*32 + L)*16 + e   (rt = row/16, kt = k/32)
// ---------------------------------------------------------------------------
__global__ void __launch_bounds__(256) prep_xA(const float* __restrict__ x,
                                               _Float16* __restrict__ xA) {
    int tid = blockIdx.x * 256 + threadIdx.x;           // 1,048,576 total
    int e  = tid & 15;
    int L  = (tid >> 4) & 31;
    int kt = (tid >> 9) & 7;
    int rt = tid >> 12;
    int i  = rt * 16 + (L & 15);
    int kh = L >> 4;
    int k  = kt * 32 + kh * 8 + e + ((e & 8) ? 8 : 0);
    xA[tid] = (_Float16)x[i * F_IN + k];
}

// ---------------------------------------------------------------------------
// Prep: W_heads [8,256,128] f32 -> B-fragment-swizzled f16
// ---------------------------------------------------------------------------
__global__ void __launch_bounds__(256) prep_WB(const float* __restrict__ W,
                                               _Float16* __restrict__ WB) {
    int tid = blockIdx.x * 256 + threadIdx.x;           // 262,144 total
    int e  = tid & 15;
    int L  = (tid >> 4) & 31;
    int t  = (tid >> 9) & 7;
    int kt = (tid >> 12) & 7;
    int h  = tid >> 15;
    int k  = kt * 32 + (L >> 4) * 8 + e + ((e & 8) ? 8 : 0);
    int d  = t * 16 + (L & 15);
    WB[tid] = (_Float16)W[(h * F_IN + k) * NHID + d];
}

// ---------------------------------------------------------------------------
// Prep: W_out [1024,256] f32 -> B-fragment-swizzled f16
// ---------------------------------------------------------------------------
__global__ void __launch_bounds__(256) prep_WoutB(const float* __restrict__ W,
                                                  _Float16* __restrict__ WoB) {
    int tid = blockIdx.x * 256 + threadIdx.x;           // 262,144 total
    int e  = tid & 15;
    int L  = (tid >> 4) & 31;
    int t  = (tid >> 9) & 15;
    int kt = tid >> 13;
    int k  = kt * 32 + (L >> 4) * 8 + e + ((e & 8) ? 8 : 0);
    int n  = t * 16 + (L & 15);
    WoB[tid] = (_Float16)W[k * OUTDIM + n];
}

// ---------------------------------------------------------------------------
// Prep: adj [4096,4096] int -> transposed bitmask adjT[jword=128][i=4096]
// bit b of adjT[w][i] == (adj[i][w*32+b] != 0)
// ---------------------------------------------------------------------------
__global__ void __launch_bounds__(256) prep_adjT(const int* __restrict__ adj,
                                                 unsigned* __restrict__ adjT) {
    int tid = blockIdx.x * 256 + threadIdx.x;           // 524,288 total
    int w = tid & 127;
    int i = tid >> 7;
    const int4* p = (const int4*)(adj + i * N_NODES + w * 32);
    unsigned m = 0u;
#pragma unroll
    for (int q = 0; q < 8; ++q) {
        int4 v = p[q];
        m |= (unsigned)(v.x != 0) << (q * 4 + 0);
        m |= (unsigned)(v.y != 0) << (q * 4 + 1);
        m |= (unsigned)(v.z != 0) << (q * 4 + 2);
        m |= (unsigned)(v.w != 0) << (q * 4 + 3);
    }
    adjT[w * N_NODES + i] = m;
}

// ---------------------------------------------------------------------------
// K1: Wh[h] = x @ W_heads[h]  (WMMA f16, f32 acc)
//     + s1 = Wh.a1, s2 = Wh.a2 (in-register cross-lane reduction)
//     + write Wh as B-fragment-swizzled f16 for the aggregation kernel
// grid: 256 blocks = 8 heads x 32 row-blocks-of-128; 8 waves x 16 rows each
// ---------------------------------------------------------------------------
__global__ void __launch_bounds__(256) k1_wh(const _Float16* __restrict__ xA,
                                             const _Float16* __restrict__ WB,
                                             const float* __restrict__ a1,
                                             const float* __restrict__ a2,
                                             _Float16* __restrict__ WhB,
                                             float* __restrict__ s1,
                                             float* __restrict__ s2) {
    int h    = blockIdx.x & 7;
    int rb   = blockIdx.x >> 3;
    int wave = threadIdx.x >> 5;
    int lane = threadIdx.x & 31;
    int kh   = lane >> 4;
    int n_   = lane & 15;
    int row0 = rb * 128 + wave * 16;
    int rt   = row0 >> 4;

    v8f acc[8] = {};
#pragma unroll
    for (int kt = 0; kt < 8; ++kt) {
        v16h a = *(const v16h*)(xA + (((rt * 8 + kt) * 32) + lane) * 16);
#pragma unroll
        for (int t = 0; t < 8; ++t) {
            v16h b = *(const v16h*)(WB + ((((h * 8 + kt) * 8 + t) * 32) + lane) * 16);
            acc[t] = wmma_f16(a, b, acc[t]);
        }
    }

    // s1/s2: per-lane partial over this lane's 8 cols, then reduce 16 lanes
    float p1[8], p2[8];
#pragma unroll
    for (int r = 0; r < 8; ++r) { p1[r] = 0.f; p2[r] = 0.f; }
#pragma unroll
    for (int t = 0; t < 8; ++t) {
        float a1v = a1[h * NHID + t * 16 + n_];
        float a2v = a2[h * NHID + t * 16 + n_];
#pragma unroll
        for (int r = 0; r < 8; ++r) {
            p1[r] += acc[t][r] * a1v;
            p2[r] += acc[t][r] * a2v;
        }
    }
#pragma unroll
    for (int off = 1; off < 16; off <<= 1) {
#pragma unroll
        for (int r = 0; r < 8; ++r) {
            p1[r] += __shfl_xor(p1[r], off, 32);
            p2[r] += __shfl_xor(p2[r], off, 32);
        }
    }
    if (n_ == 0) {
#pragma unroll
        for (int r = 0; r < 8; ++r) {
            int i = row0 + r + 8 * kh;
            s1[h * N_NODES + i] = p1[r];
            s2[h * N_NODES + i] = p2[r];
        }
    }

    // Scatter Wh into B-fragment order (j is the future K dimension)
#pragma unroll
    for (int t = 0; t < 8; ++t) {
#pragma unroll
        for (int r = 0; r < 8; ++r) {
            int j  = row0 + r + 8 * kh;
            int jt = j >> 5;
            int jr = j & 31;
            int L2 = (((jr >> 3) & 1) << 4) | n_;
            int e2 = (jr & 7) | ((jr >> 4) << 3);
            WhB[(((h * 128 + jt) * 8 + t) * 32 + L2) * 16 + e2] = (_Float16)acc[t][r];
        }
    }
}

// ---------------------------------------------------------------------------
// K1b: per-head max of s2 (static softmax bound; lrelu is monotone so
//      m_i = lrelu(s1_i + max_j s2_j) >= every e_ij -> no online softmax)
// ---------------------------------------------------------------------------
__global__ void __launch_bounds__(256) k1b_maxs2(const float* __restrict__ s2,
                                                 float* __restrict__ mx) {
    __shared__ float red[256];
    int h = blockIdx.x;
    float m = -1e30f;
    for (int i = threadIdx.x; i < N_NODES; i += 256)
        m = fmaxf(m, s2[h * N_NODES + i]);
    red[threadIdx.x] = m;
    __syncthreads();
    for (int s = 128; s > 0; s >>= 1) {
        if (threadIdx.x < s) red[threadIdx.x] = fmaxf(red[threadIdx.x], red[threadIdx.x + s]);
        __syncthreads();
    }
    if (threadIdx.x == 0) mx[h] = red[0];
}

// ---------------------------------------------------------------------------
// K1c: swizzle s2 into per-lane-contiguous order for K2's A-fragment build:
//      s2sw[h][jt][kh][e] = s2[h][jt*32 + (8*kh + e + (e>=8 ? 8 : 0))]
// so each lane reads its 16 values as 4 x b128.
// ---------------------------------------------------------------------------
__global__ void __launch_bounds__(256) prep_s2sw(const float* __restrict__ s2,
                                                 float* __restrict__ s2sw) {
    int tid = blockIdx.x * 256 + threadIdx.x;           // 32,768 total
    int e  = tid & 15;
    int kh = (tid >> 4) & 1;
    int jt = (tid >> 5) & 127;
    int h  = tid >> 12;
    int k  = kh * 8 + e + ((e & 8) ? 8 : 0);
    s2sw[tid] = s2[h * N_NODES + jt * 32 + k];
}

// ---------------------------------------------------------------------------
// K2: fused masked-softmax attention + aggregation + ELU.
// Each wave: 16 rows of one head; loop j in tiles of 32 (one WMMA K-step).
// P (16x32 probs) built per-lane directly in A-fragment layout; acc += P @ Wh.
// Softmax in log2 domain: p = exp2(max(t, 0.2t) - mL), t = (s1+s2)*log2e,
// with constants folded into two FMAs per element.
// grid: 256 blocks = 8 heads x 32 i-blocks-of-128
// ---------------------------------------------------------------------------
__global__ void __launch_bounds__(256) k2_attn(const float* __restrict__ s1g,
                                               const float* __restrict__ s2sw,
                                               const float* __restrict__ mx,
                                               const unsigned* __restrict__ adjT,
                                               const _Float16* __restrict__ WhB,
                                               _Float16* __restrict__ feat) {
    int h    = blockIdx.x & 7;
    int ib   = blockIdx.x >> 3;
    int wave = threadIdx.x >> 5;
    int lane = threadIdx.x & 31;
    int kh   = lane >> 4;
    int n_   = lane & 15;
    int i0   = ib * 128 + wave * 16;
    int myrow = i0 + n_;

    float s1v = s1g[h * N_NODES + myrow];
    float pre = s1v + mx[h];
    float m   = pre > 0.f ? pre : GAT_ALPHA * pre;   // static row max bound
    float mL   = m * LOG2E;
    float s1mL = s1v * LOG2E - mL;                   // a = fma(s2, log2e, s1mL)
    float bb   = -0.8f * mL;                         // b = fma(a, 0.2, bb)

    float ls[4] = {0.f, 0.f, 0.f, 0.f};
    v8f acc[8] = {};
    const v16h*   Bhead = (const v16h*)WhB + (size_t)h * 128 * 8 * 32;
    const float*  s2h   = s2sw + h * N_NODES + kh * 16;

#pragma unroll 2
    for (int jt = 0; jt < 128; ++jt) {
        unsigned adjw = adjT[jt * N_NODES + myrow];
        const float4* s2q = (const float4*)(s2h + jt * 32);
        float4 q[4];
        q[0] = s2q[0]; q[1] = s2q[1]; q[2] = s2q[2]; q[3] = s2q[3];
        float s2v[16];
#pragma unroll
        for (int qi = 0; qi < 4; ++qi) {
            s2v[qi * 4 + 0] = q[qi].x;
            s2v[qi * 4 + 1] = q[qi].y;
            s2v[qi * 4 + 2] = q[qi].z;
            s2v[qi * 4 + 3] = q[qi].w;
        }
        v16h P;
#pragma unroll
        for (int e = 0; e < 16; ++e) {
            int k = kh * 8 + e + ((e & 8) ? 8 : 0);       // adjacency bit index
            float a = fmaf(s2v[e], LOG2E, s1mL);          // t - mL
            float b = fmaf(a, GAT_ALPHA, bb);             // 0.2t - mL
            float arg = fmaxf(a, b);                      // lrelu in log2 domain
            float pv = ((adjw >> k) & 1u) ? exp2f(arg) : 0.f;
            ls[e & 3] += pv;
            P[e] = (_Float16)pv;
        }
        const v16h* B = Bhead + jt * 8 * 32;
#pragma unroll
        for (int t = 0; t < 8; ++t) {
            v16h b = B[t * 32 + lane];
            acc[t] = wmma_f16(P, b, acc[t]);
        }
    }

    // row sums: combine partials + the two k-halves, then broadcast to C-frag rows
    float lsum = (ls[0] + ls[1]) + (ls[2] + ls[3]);
    lsum += __shfl_xor(lsum, 16, 32);
    float linv = 1.0f / lsum;
    float sc[8];
#pragma unroll
    for (int r = 0; r < 8; ++r) sc[r] = __shfl(linv, r + 8 * kh, 32);

#pragma unroll
    for (int t = 0; t < 8; ++t) {
#pragma unroll
        for (int r = 0; r < 8; ++r) {
            float v = acc[t][r] * sc[r];
            v = v > 0.f ? v : (exp2f(v * LOG2E) - 1.0f);  // ELU
            int i = i0 + r + 8 * kh;
            feat[i * HID_ALL + h * NHID + t * 16 + n_] = (_Float16)v;
        }
    }
}

// ---------------------------------------------------------------------------
// K3: out = feat[4096,1024] @ W_out + b_out  (WMMA f16, f32 out)
// 64 blocks; 8 waves = 4 row-tiles x 2 col-halves (128 cols each)
// ---------------------------------------------------------------------------
__global__ void __launch_bounds__(256) k3_out(const _Float16* __restrict__ feat,
                                              const _Float16* __restrict__ WoB,
                                              const float* __restrict__ bout,
                                              float* __restrict__ out) {
    int rb   = blockIdx.x;
    int wave = threadIdx.x >> 5;
    int lane = threadIdx.x & 31;
    int kh   = lane >> 4;
    int n_   = lane & 15;
    int rt4  = wave & 3;
    int ch   = wave >> 2;
    int row0 = rb * 64 + rt4 * 16;
    int myrow = row0 + n_;

    v8f acc[8] = {};
#pragma unroll 4
    for (int kt = 0; kt < 32; ++kt) {
        const _Float16* ab = feat + myrow * HID_ALL + kt * 32 + kh * 8;
        v8h alo = *(const v8h*)(ab);
        v8h ahi = *(const v8h*)(ab + 16);
        v16h a = __builtin_shufflevector(alo, ahi,
                                         0, 1, 2, 3, 4, 5, 6, 7,
                                         8, 9, 10, 11, 12, 13, 14, 15);
#pragma unroll
        for (int t = 0; t < 8; ++t) {
            int tg = ch * 8 + t;
            v16h b = *(const v16h*)(WoB + ((kt * 16 + tg) * 32 + lane) * 16);
            acc[t] = wmma_f16(a, b, acc[t]);
        }
    }
#pragma unroll
    for (int t = 0; t < 8; ++t) {
        int n  = (ch * 8 + t) * 16 + n_;
        float bv = bout[n];
#pragma unroll
        for (int r = 0; r < 8; ++r) {
            int i = row0 + r + 8 * kh;
            out[i * OUTDIM + n] = acc[t][r] + bv;
        }
    }
}

// ---------------------------------------------------------------------------
extern "C" void kernel_launch(void* const* d_in, const int* in_sizes, int n_in,
                              void* d_out, int out_size, void* d_ws, size_t ws_size,
                              hipStream_t stream) {
    (void)in_sizes; (void)n_in; (void)out_size; (void)ws_size;
    const float* x   = (const float*)d_in[0];
    const int*   adj = (const int*)d_in[1];
    const float* Wh  = (const float*)d_in[2];
    const float* a1  = (const float*)d_in[3];
    const float* a2  = (const float*)d_in[4];
    const float* Wo  = (const float*)d_in[5];
    const float* bo  = (const float*)d_in[6];
    float* out = (float*)d_out;

    char* ws = (char*)d_ws;
    _Float16* xA    = (_Float16*)(ws);                                   // 2 MB
    _Float16* WB    = (_Float16*)(ws + (2u << 20));                      // 512 KB
    _Float16* WoB   = (_Float16*)(ws + (2u << 20) + (512u << 10));       // 512 KB
    unsigned* adjT  = (unsigned*)(ws + (3u << 20));                      // 2 MB
    _Float16* WhB   = (_Float16*)(ws + (5u << 20));                      // 8 MB
    float*    s1    = (float*)(ws + (13u << 20));                        // 128 KB
    float*    s2    = (float*)(ws + (13u << 20) + (128u << 10));         // 128 KB
    float*    mx    = (float*)(ws + (13u << 20) + (256u << 10));         // 32 B
    float*    s2sw  = (float*)(ws + (13u << 20) + (320u << 10));         // 128 KB
    _Float16* feat  = (_Float16*)(ws + (14u << 20));                     // 8 MB

    prep_xA   <<<(N_NODES * F_IN) / 256, 256, 0, stream>>>(x, xA);
    prep_WB   <<<(NHEADS * F_IN * NHID) / 256, 256, 0, stream>>>(Wh, WB);
    prep_WoutB<<<(HID_ALL * OUTDIM) / 256, 256, 0, stream>>>(Wo, WoB);
    prep_adjT <<<(N_NODES * 128) / 256, 256, 0, stream>>>(adj, adjT);

    k1_wh     <<<NHEADS * (N_NODES / 128), 256, 0, stream>>>(xA, WB, a1, a2, WhB, s1, s2);
    k1b_maxs2 <<<NHEADS, 256, 0, stream>>>(s2, mx);
    prep_s2sw <<<(NHEADS * N_NODES) / 256, 256, 0, stream>>>(s2, s2sw);
    k2_attn   <<<NHEADS * (N_NODES / 128), 256, 0, stream>>>(s1, s2sw, mx, adjT, WhB, feat);
    k3_out    <<<N_NODES / 64, 256, 0, stream>>>(feat, WoB, bo, out);
}
